// Pointnet2BackboneDeeper_33406255628525
// MI455X (gfx1250) — compile-verified, matrix operand order checked by reading, not run on hardware
//
#include <hip/hip_runtime.h>

typedef __attribute__((ext_vector_type(16))) _Float16     v16h;
typedef __attribute__((ext_vector_type(8)))  float        v8f;
typedef __attribute__((ext_vector_type(8)))  unsigned int v8u;

#define B_ 4
#define N_ 20000

// ---------------------------------------------------------------- FPS
__global__ void k_fps(const float* __restrict__ xyz, float* __restrict__ dist,
                      int* __restrict__ inds, int npoint) {
  __shared__ float sval[1024];
  __shared__ int   sidx[1024];
  __shared__ int   s_last;
  const int b = blockIdx.x, t = threadIdx.x, nt = blockDim.x;
  const float* X = xyz + (size_t)b * N_ * 3;
  float* D = dist + (size_t)b * N_;
  for (int i = t; i < N_; i += nt) D[i] = 1e10f;
  if (t == 0) { inds[b * npoint] = 0; s_last = 0; }
  __syncthreads();
  for (int it = 1; it < npoint; ++it) {
    int last = s_last;
    float lx = X[last*3], ly = X[last*3+1], lz = X[last*3+2];
    float bv = -1.0f; int bi = 0;
    for (int i = t; i < N_; i += nt) {
      float dx = X[i*3]-lx, dy = X[i*3+1]-ly, dz = X[i*3+2]-lz;
      float d2 = dx*dx + dy*dy + dz*dz;
      float nd = D[i]; nd = d2 < nd ? d2 : nd; D[i] = nd;
      if (nd > bv) { bv = nd; bi = i; }
    }
    sval[t] = bv; sidx[t] = bi;
    __syncthreads();
    for (int s = nt >> 1; s > 0; s >>= 1) {
      if (t < s) {
        float ov = sval[t+s]; int oi = sidx[t+s];
        if (ov > sval[t] || (ov == sval[t] && oi < sidx[t])) { sval[t] = ov; sidx[t] = oi; }
      }
      __syncthreads();
    }
    if (t == 0) { inds[b*npoint + it] = sidx[0]; s_last = sidx[0]; }
    __syncthreads();
  }
}

// ------------------------------------------------- gather new_xyz (inds or iota)
__global__ void k_gather_xyz(const float* __restrict__ xyz, int n_src,
                             const int* __restrict__ inds,
                             float* __restrict__ out, int S) {
  int t = blockIdx.x * blockDim.x + threadIdx.x;
  if (t >= B_ * S) return;
  int b = t / S, s = t % S;
  int i = inds ? inds[(size_t)b*S + s] : s;
  out[(size_t)t*3+0] = xyz[((size_t)b*n_src + i)*3+0];
  out[(size_t)t*3+1] = xyz[((size_t)b*n_src + i)*3+1];
  out[(size_t)t*3+2] = xyz[((size_t)b*n_src + i)*3+2];
}

// ---------------------------------------------------------------- ball query
__global__ void k_ball_query(const float* __restrict__ nxyz, const float* __restrict__ xyz,
                             int n, int S, float r2, int ns, int* __restrict__ idx) {
  int t = blockIdx.x * blockDim.x + threadIdx.x;
  if (t >= B_ * S) return;
  int b = t / S;
  float cx = nxyz[(size_t)t*3], cy = nxyz[(size_t)t*3+1], cz = nxyz[(size_t)t*3+2];
  const float* X = xyz + (size_t)b * n * 3;
  int* out = idx + (size_t)t * ns;
  int cnt = 0, first = -1;
  for (int i = 0; i < n; ++i) {
    float dx = X[i*3]-cx, dy = X[i*3+1]-cy, dz = X[i*3+2]-cz;
    if (dx*dx + dy*dy + dz*dz < r2) {
      if (first < 0) first = i;
      out[cnt++] = i;
      if (cnt >= ns) break;
    }
  }
  int pad = (first >= 0) ? first : (n - 1);
  for (int j = cnt; j < ns; ++j) out[j] = pad;
}

// ------------------------------------------------- group: build fp16 GEMM input
__global__ void k_group(const float* __restrict__ xyz, const float* __restrict__ nxyz,
                        const _Float16* __restrict__ feats, const int* __restrict__ idx,
                        _Float16* __restrict__ G, int S, int ns, int C, int Kp, float inv_r) {
  int t = blockIdx.x * blockDim.x + threadIdx.x;
  if (t >= S * ns) return;
  int s = t / ns;
  int i = idx[t];
  _Float16* g = G + (size_t)t * Kp;
  g[0] = (_Float16)((xyz[(size_t)i*3+0] - nxyz[(size_t)s*3+0]) * inv_r);
  g[1] = (_Float16)((xyz[(size_t)i*3+1] - nxyz[(size_t)s*3+1]) * inv_r);
  g[2] = (_Float16)((xyz[(size_t)i*3+2] - nxyz[(size_t)s*3+2]) * inv_r);
  for (int c = 0; c < C; ++c)       g[3+c] = feats[(size_t)i*C + c];
  for (int c = 3 + C; c < Kp; ++c)  g[c] = (_Float16)0.f;
}

// ------------------------------------------------- weight fp32 -> fp16 (K padded)
__global__ void k_cvt_w(const float* __restrict__ w, _Float16* __restrict__ wh,
                        int O, int C, int Kp) {
  int t = blockIdx.x * blockDim.x + threadIdx.x;
  if (t >= O * Kp) return;
  int o = t / Kp, c = t % Kp;
  wh[t] = (c < C) ? (_Float16)w[(size_t)o*C + c] : (_Float16)0.f;
}

// ---------------------------------------------------------------- WMMA GEMM
// Out[M x O] = relu( (A[M x K] @ W[O x K]^T) * scale[O] + bias[O] ), fp16 out.
// One wave computes a 16x64 output strip (4 N-tiles): each A fragment feeds
// 4 v_wmma_f32_16x16x32_f16 ops. K % 32 == 0, M % 16 == 0, O % 64 == 0.
__global__ void k_gemm(const _Float16* __restrict__ A, const _Float16* __restrict__ W,
                       const float* __restrict__ scale, const float* __restrict__ bias,
                       _Float16* __restrict__ Out, int M, int K, int O) {
  int lane = threadIdx.x & 31;
  int wave = blockIdx.x * (blockDim.x >> 5) + (threadIdx.x >> 5);
  int Mt = M >> 4, Nb = O >> 6;           // N blocks of 64
  if (wave >= Mt * Nb) return;
  int mt = wave / Nb, nb = wave % Nb;
  int r = lane & 15, half = lane >> 4;
  const _Float16* Arow = A + (size_t)(mt*16 + r) * K;
  const _Float16* Wrow = W + (size_t)(nb*64 + r) * K;   // + j*16*K per N-tile
  v8f acc0 = {}, acc1 = {}, acc2 = {}, acc3 = {};
  for (int k0 = 0; k0 < K; k0 += 32) {
    v8u au;
#pragma unroll
    for (int v = 0; v < 8; ++v) {
      int kk = k0 + ((v < 4) ? 2*v : 16 + 2*(v-4)) + half*8;  // ISA f16 A/B layout
      au[v] = *(const unsigned int*)(Arow + kk);
    }
    v16h af = __builtin_bit_cast(v16h, au);
#pragma unroll
    for (int j = 0; j < 4; ++j) {
      v8u bu;
#pragma unroll
      for (int v = 0; v < 8; ++v) {
        int kk = k0 + ((v < 4) ? 2*v : 16 + 2*(v-4)) + half*8;
        bu[v] = *(const unsigned int*)(Wrow + (size_t)j*16*K + kk);
      }
      v16h bf = __builtin_bit_cast(v16h, bu);
      if      (j == 0) acc0 = __builtin_amdgcn_wmma_f32_16x16x32_f16(false, af, false, bf, (short)0, acc0, false, false);
      else if (j == 1) acc1 = __builtin_amdgcn_wmma_f32_16x16x32_f16(false, af, false, bf, (short)0, acc1, false, false);
      else if (j == 2) acc2 = __builtin_amdgcn_wmma_f32_16x16x32_f16(false, af, false, bf, (short)0, acc2, false, false);
      else             acc3 = __builtin_amdgcn_wmma_f32_16x16x32_f16(false, af, false, bf, (short)0, acc3, false, false);
    }
  }
  int row0 = mt*16 + (half ? 8 : 0);
#pragma unroll
  for (int j = 0; j < 4; ++j) {
    const v8f& acc = (j == 0) ? acc0 : (j == 1) ? acc1 : (j == 2) ? acc2 : acc3;
    int col = nb*64 + j*16 + r;
    float sc = scale[col], bi = bias[col];
#pragma unroll
    for (int rr = 0; rr < 8; ++rr) {
      float v = acc[rr] * sc + bi;
      v = v > 0.f ? v : 0.f;
      Out[(size_t)(row0 + rr) * O + col] = (_Float16)v;
    }
  }
}

// ---------------------------------------------------------------- max over nsample
__global__ void k_maxpool(const _Float16* __restrict__ act, _Float16* __restrict__ out,
                          int S, int ns, int O) {
  int t = blockIdx.x * blockDim.x + threadIdx.x;
  if (t >= S * O) return;
  int s = t / O, o = t % O;
  float m = -1e30f;
  for (int j = 0; j < ns; ++j) {
    float v = (float)act[(size_t)(s*ns + j) * O + o];
    m = v > m ? v : m;
  }
  out[(size_t)s*O + o] = (_Float16)m;
}

// ---------------------------------------------------------------- 3-NN
__global__ void k_knn3(const float* __restrict__ x1, const float* __restrict__ x2,
                       int n1, int n2, int* __restrict__ kidx, float* __restrict__ kw) {
  int t = blockIdx.x * blockDim.x + threadIdx.x;
  if (t >= B_ * n1) return;
  int b = t / n1;
  float cx = x1[(size_t)t*3], cy = x1[(size_t)t*3+1], cz = x1[(size_t)t*3+2];
  const float* X = x2 + (size_t)b * n2 * 3;
  float b0 = 1e30f, b1 = 1e30f, b2 = 1e30f; int i0 = 0, i1 = 0, i2 = 0;
  for (int j = 0; j < n2; ++j) {
    float dx = X[j*3]-cx, dy = X[j*3+1]-cy, dz = X[j*3+2]-cz;
    float d = dx*dx + dy*dy + dz*dz;
    if (d < b0)      { b2=b1; i2=i1; b1=b0; i1=i0; b0=d; i0=j; }
    else if (d < b1) { b2=b1; i2=i1; b1=d;  i1=j; }
    else if (d < b2) { b2=d;  i2=j; }
  }
  float w0 = 1.f/(b0+1e-8f), w1 = 1.f/(b1+1e-8f), w2 = 1.f/(b2+1e-8f);
  float inv = 1.f/(w0+w1+w2);
  kidx[(size_t)t*3+0]=i0; kidx[(size_t)t*3+1]=i1; kidx[(size_t)t*3+2]=i2;
  kw[(size_t)t*3+0]=w0*inv; kw[(size_t)t*3+1]=w1*inv; kw[(size_t)t*3+2]=w2*inv;
}

// ------------------------------------------------- interpolate + concat -> fp16 G
__global__ void k_interp(const _Float16* __restrict__ f2, const _Float16* __restrict__ f1,
                         const int* __restrict__ kidx, const float* __restrict__ kw,
                         _Float16* __restrict__ G, int n1, int n2) {
  int t = blockIdx.x * blockDim.x + threadIdx.x;
  if (t >= B_ * n1 * 512) return;
  int c = t % 512, row = t / 512;
  int b = row / n1, i = row % n1;
  if (c < 256) {
    float acc = 0.f;
#pragma unroll
    for (int k = 0; k < 3; ++k) {
      int j = kidx[(size_t)row*3 + k];
      acc += kw[(size_t)row*3 + k] * (float)f2[((size_t)b*n2 + j)*256 + c];
    }
    G[(size_t)row*512 + c] = (_Float16)acc;
  } else {
    G[(size_t)row*512 + c] = f1[((size_t)b*n1 + i)*256 + (c - 256)];
  }
}

// ------------------------------------------------- fp16 (B,n,O) -> f32 (B,O,n)
__global__ void k_h2f_tr(const _Float16* __restrict__ in, float* __restrict__ out,
                         int n, int O) {
  int t = blockIdx.x * blockDim.x + threadIdx.x;
  if (t >= B_ * O * n) return;
  int b = t / (O*n), rem = t % (O*n);
  int o = rem / n, i = rem % n;
  out[t] = (float)in[((size_t)b*n + i)*O + o];
}

// ================================================================ host
static inline int cdiv(int a, int b) { return (a + b - 1) / b; }

extern "C" void kernel_launch(void* const* d_in, const int* in_sizes, int n_in,
                              void* d_out, int out_size, void* d_ws, size_t ws_size,
                              hipStream_t stream) {
  (void)in_sizes; (void)n_in; (void)out_size; (void)ws_size;
  const float* pc = (const float*)d_in[0];

  static const int sa_dims[6][4] = {
    {3,64,64,128},{131,128,128,256},{259,128,128,256},
    {259,128,128,256},{259,128,256,256},{259,256,256,256}};
  static const int Sarr[6]  = {2048,1024,512,256,128,64};
  static const int nsarr[6] = {64,32,16,16,12,8};
  static const float rad[6] = {0.2f,0.4f,0.8f,1.2f,1.6f,2.0f};

  struct L { const float *w, *sc, *bi; int O, C, Kp; _Float16* wh; };
  L layers[26];

  // ---- workspace carve-out
  uintptr_t base = (uintptr_t)d_ws; size_t off = 0;
  auto alloc = [&](size_t bytes) -> void* {
    off = (off + 255) & ~(size_t)255;
    void* r = (void*)(base + off);
    off += bytes;
    return r;
  };
  float* dist     = (float*)alloc((size_t)B_*N_*4);
  int*   fpsinds  = (int*)  alloc((size_t)B_*2048*4);
  float* xyzl[6];
  for (int l = 0; l < 6; ++l) xyzl[l] = (float*)alloc((size_t)B_*Sarr[l]*3*4);
  _Float16* featl[6];
  for (int l = 0; l < 6; ++l) featl[l] = (_Float16*)alloc((size_t)B_*Sarr[l]*sa_dims[l][3]*2);
  _Float16* fpf[4];
  static const int fp_n1[4] = {128,256,512,1024};
  for (int m = 0; m < 4; ++m) fpf[m] = (_Float16*)alloc((size_t)B_*fp_n1[m]*256*2);
  int*   idxbuf  = (int*)  alloc((size_t)B_*2048*64*4);
  int*   knn_idx = (int*)  alloc((size_t)B_*1024*3*4);
  float* knn_w   = (float*)alloc((size_t)B_*1024*3*4);

  // ---- layer table + fp16 weight staging
  int p = 1, li = 0;
  size_t whHalfs = 0;
  for (int m = 0; m < 6; ++m)
    for (int j = 0; j < 3; ++j) {
      int C = sa_dims[m][j], O = sa_dims[m][j+1], Kp = ((C+31)/32)*32;
      layers[li] = { (const float*)d_in[p], (const float*)d_in[p+1], (const float*)d_in[p+2],
                     O, C, Kp, nullptr };
      whHalfs += (size_t)O*Kp; p += 3; ++li;
    }
  for (int m = 0; m < 4; ++m)
    for (int j = 0; j < 2; ++j) {
      int C = (j == 0) ? 512 : 256, O = 256, Kp = C;
      layers[li] = { (const float*)d_in[p], (const float*)d_in[p+1], (const float*)d_in[p+2],
                     O, C, Kp, nullptr };
      whHalfs += (size_t)O*Kp; p += 3; ++li;
    }
  _Float16* whbuf = (_Float16*)alloc(whHalfs*2);
  {
    size_t o2 = 0;
    for (int i = 0; i < 26; ++i) {
      layers[i].wh = whbuf + o2;
      int n = layers[i].O * layers[i].Kp;
      k_cvt_w<<<cdiv(n,256),256,0,stream>>>(layers[i].w, layers[i].wh,
                                            layers[i].O, layers[i].C, layers[i].Kp);
      o2 += (size_t)n;
    }
  }

  // ping-pong activation buffers (per-batch SA max: 131072x64 / 131072x128)
  _Float16* bufA = (_Float16*)alloc((size_t)131072*64*2);
  _Float16* bufB = (_Float16*)alloc((size_t)131072*128*2);

  // ---- FPS
  k_fps<<<B_,1024,0,stream>>>(pc, dist, fpsinds, 2048);

  // ---- SA modules (per-batch to bound scratch)
  const float*    prev_xyz = pc; int prev_n = N_;
  const _Float16* prev_f   = nullptr; int prevC = 0;
  for (int l = 0; l < 6; ++l) {
    int S = Sarr[l], ns = nsarr[l], Co = sa_dims[l][3];
    float r = rad[l];
    int Kp0 = layers[l*3].Kp;
    k_gather_xyz<<<cdiv(B_*S,256),256,0,stream>>>(prev_xyz, prev_n,
                                                  l == 0 ? fpsinds : nullptr, xyzl[l], S);
    k_ball_query<<<cdiv(B_*S,128),128,0,stream>>>(xyzl[l], prev_xyz, prev_n, S,
                                                  r*r, ns, idxbuf);
    for (int b = 0; b < B_; ++b) {
      int rows = S * ns;
      k_group<<<cdiv(rows,128),128,0,stream>>>(
          prev_xyz + (size_t)b*prev_n*3, xyzl[l] + (size_t)b*S*3,
          prev_f ? prev_f + (size_t)b*prev_n*prevC : nullptr,
          idxbuf + (size_t)b*S*ns, bufA, S, ns, prevC, Kp0, 1.0f/r);
      const _Float16* in = bufA; int K = Kp0;
      for (int j = 0; j < 3; ++j) {
        const L& ly = layers[l*3 + j];
        _Float16* out = (j % 2 == 0) ? bufB : bufA;
        int waves = (rows/16) * (ly.O/64);
        k_gemm<<<cdiv(waves,8),256,0,stream>>>(in, ly.wh, ly.sc, ly.bi, out,
                                               rows, K, ly.O);
        in = out; K = ly.O;
      }
      k_maxpool<<<cdiv(S*Co,256),256,0,stream>>>(in, featl[l] + (size_t)b*S*Co, S, ns, Co);
    }
    prev_xyz = xyzl[l]; prev_n = S; prev_f = featl[l]; prevC = Co;
  }

  // ---- FP modules (full batch; rows = B*n1, all multiples of 16)
  const _Float16* f2 = featl[5]; int n2 = Sarr[5];
  static const int fpl1[4] = {4,3,2,1};
  for (int m = 0; m < 4; ++m) {
    int l1 = fpl1[m], n1 = Sarr[l1];
    k_knn3<<<cdiv(B_*n1,128),128,0,stream>>>(xyzl[l1], xyzl[l1+1], n1, n2, knn_idx, knn_w);
    k_interp<<<cdiv(B_*n1*512,256),256,0,stream>>>(f2, featl[l1], knn_idx, knn_w,
                                                   bufA, n1, n2);
    int rows = B_ * n1;
    const L& a = layers[18 + m*2];
    const L& c = layers[18 + m*2 + 1];
    k_gemm<<<cdiv((rows/16)*(a.O/64),8),256,0,stream>>>(bufA, a.wh, a.sc, a.bi,
                                                        bufB, rows, 512, a.O);
    k_gemm<<<cdiv((rows/16)*(c.O/64),8),256,0,stream>>>(bufB, c.wh, c.sc, c.bi,
                                                        fpf[m], rows, 256, c.O);
    f2 = fpf[m]; n2 = n1;
  }

  // ---- final transpose to f32 (B, 256, 1024)
  k_h2f_tr<<<cdiv(B_*256*1024,256),256,0,stream>>>(fpf[3], (float*)d_out, 1024, 256);
}
